// ScaleShiftMACE_43576738185433
// MI455X (gfx1250) — matrix operand order repaired and checked by reading, not compiled, not run on hardware
//
#include <hip/hip_runtime.h>
#include <math.h>

// ============================================================================
// MACE-style GNN forward for MI455X (gfx1250, wave32).
// Dense GEMMs (edge MLPs, node channel-mixing linears) run on the matrix pipe
// via v_wmma_f32_16x16x32_f16 (f16 inputs, f32 accumulate). All WMMA B
// operands (weights) are re-packed once per launch into fragment-major layout
// so each lane loads its 32-byte fragment with 2x global_load_b128 instead of
// 16 strided d16 loads. Message scatter uses global_atomic_add_f32.
// Element-dependent (one-hot) einsums are species-indexed per-node matvecs on
// the VALU (weights are L2-resident). The vector message of interaction 2
// (m2_v, incl. the cross product) is dead code in the reference -> dropped,
// and only the 256 live columns of mlp2's 640-wide output layer are packed.
// Workspace use: ~143 MB.
// ============================================================================

typedef _Float16 v16h __attribute__((ext_vector_type(16)));
typedef float    v8f  __attribute__((ext_vector_type(8)));

#define NUM_EL     10
#define CCH        128
#define RMAXF      5.0f
#define AVG_INV    (1.0f / 10.0f)
#define SQRT3F     1.7320508075688772f
#define INV_SQRT3F (1.0f / 1.7320508075688772f)
#define EP_WAVES   2

__device__ __forceinline__ float silu_f(float x) { return x / (1.0f + __expf(-x)); }

__device__ __forceinline__ v8f wmma16(v16h a, v16h b, v8f c) {
  // D = A(16x32 f16) * B(32x16 f16) + C(16x16 f32)
  return __builtin_amdgcn_wmma_f32_16x16x32_f16(false, a, false, b, (short)0, c,
                                                false, false);
}

// A fragment from an LDS tile laid out [16 rows][64 cols] f16.
// ISA 7.12.2: lane<16 -> K = k0+{0..7, 16..23}; lane>=16 -> K = k0+{8..15, 24..31}
__device__ __forceinline__ v16h load_a_lds64(const _Float16* st, int lane, int k0) {
  const int row = lane & 15;
  const int kb  = k0 + ((lane & 16) ? 8 : 0);
  v16h a;
#pragma unroll
  for (int i = 0; i < 8; ++i) a[i] = st[row * 64 + kb + i];
#pragma unroll
  for (int i = 0; i < 8; ++i) a[8 + i] = st[row * 64 + kb + 16 + i];
  return a;
}

// B fragment from a fragment-major packed weight: P[tks][lane][16 halves].
// tks = ntile * ksteps + kstep. 32B contiguous per lane -> 2x b128 loads.
__device__ __forceinline__ v16h load_bp(const _Float16* P, int tks, int lane) {
  return *(const v16h*)(P + (((size_t)tks * 32 + lane) << 4));
}

// One 64->64 MLP hidden layer for a 16-edge tile: stOut = silu(stIn @ w)
// w is fragment-major packed with ksteps = 2, ntiles = 4.
__device__ __forceinline__ void mlp_hidden_layer(const _Float16* stIn,
                                                 _Float16* stOut,
                                                 const _Float16* w, int lane) {
  const v16h a0 = load_a_lds64(stIn, lane, 0);
  const v16h a1 = load_a_lds64(stIn, lane, 32);
  const int rbase = (lane & 16) ? 8 : 0;
  const int cl = lane & 15;
#pragma unroll
  for (int t = 0; t < 4; ++t) {
    const int col = t * 16 + cl;
    v8f acc = {};
    acc = wmma16(a0, load_bp(w, t * 2 + 0, lane), acc);
    acc = wmma16(a1, load_bp(w, t * 2 + 1, lane), acc);
#pragma unroll
    for (int r = 0; r < 8; ++r)  // C/D layout: M = r (+8 for lanes 16..31)
      stOut[(size_t)(rbase + r) * 64 + col] = (_Float16)silu_f(acc[r]);
  }
}

// ---------------------------------------------------------------------------
// Pack a row-major f32 weight [K][srcStride] (first Ncols columns) into
// fragment-major f16: P[((nt*ksteps + ks)*32 + lane)*16 + i] where
// col = nt*16 + (lane&15), krow = ks*32 + ((lane&16)?8:0) + i + ((i&8)?8:0).
// ---------------------------------------------------------------------------
__global__ void pack_b_kernel(const float* __restrict__ W,
                              _Float16* __restrict__ P, int K, int Ncols,
                              int srcStride) {
  const int o = blockIdx.x * blockDim.x + threadIdx.x;
  if (o >= K * Ncols) return;
  const int i      = o & 15;
  const int lane   = (o >> 4) & 31;
  const int ksteps = K >> 5;
  const int ks     = (o >> 9) % ksteps;
  const int nt     = (o >> 9) / ksteps;
  const int col    = nt * 16 + (lane & 15);
  const int krow   = ks * 32 + ((lane & 16) ? 8 : 0) + i + ((i & 8) ? 8 : 0);
  P[o] = (_Float16)W[(size_t)krow * srcStride + col];
}

// ---------------------------------------------------------------------------
// Generic WMMA GEMM: C[M,Nn] = A[M,K](f32, row-major) @ B(packed f16).
// One wave per 16x16 output tile. M, Nn, K multiples of 16/16/32.
// ---------------------------------------------------------------------------
__global__ void wmma_gemm_kernel(const float* __restrict__ A,
                                 const _Float16* __restrict__ B,
                                 float* __restrict__ C, int M, int Nn, int K) {
  const int lane   = threadIdx.x & 31;
  const int wave   = blockIdx.x * (blockDim.x >> 5) + (threadIdx.x >> 5);
  const int ntiles = Nn >> 4;
  const int mt     = wave / ntiles;
  const int nt     = wave - mt * ntiles;
  if (mt * 16 >= M) return;

  const int rowA   = mt * 16 + (lane & 15);
  const int colC   = nt * 16 + (lane & 15);
  const int aoff   = (lane & 16) ? 8 : 0;
  const int ksteps = K >> 5;
  v8f acc = {};
  for (int ks = 0; ks < ksteps; ++ks) {
    v16h a;
    const float* ap = A + (size_t)rowA * K + (ks << 5) + aoff;
#pragma unroll
    for (int i = 0; i < 8; ++i) a[i] = (_Float16)ap[i];        // 32B contig
#pragma unroll
    for (int i = 0; i < 8; ++i) a[8 + i] = (_Float16)ap[16 + i];  // 32B contig
    acc = wmma16(a, load_bp(B, nt * ksteps + ks, lane), acc);
  }
  const int rowC = mt * 16 + ((lane & 16) ? 8 : 0);
#pragma unroll
  for (int r = 0; r < 8; ++r)
    C[(size_t)(rowC + r) * Nn + colC] = acc[r];
}

// ---------------------------------------------------------------------------
// Fused edge pass: per-wave 16-edge tile.
//   radial-MLP (8->64->64->64->256) with WMMA -> elementwise message
//   formation with gathered sender features   -> atomic scatter to nodes.
// PASS==1: out cols 0..127 -> m_s = tw0*xs ; 128..255 -> m_v = tw1*xs (x) Y
// PASS==2: out cols 0..127 -> tw0*xs2 ; 128..255 -> tw1*dot(xv2,Y)/sqrt3
// All MLP weights fragment-major packed (w1/w2: 4x2 frags, w3: 16x2 frags).
// ---------------------------------------------------------------------------
template <int PASS>
__global__ void __launch_bounds__(32 * EP_WAVES)
edge_pass_kernel(const float* __restrict__ feats, const float* __restrict__ Yg,
                 const int* __restrict__ snd, const int* __restrict__ rcv,
                 const float* __restrict__ Xs, const float* __restrict__ Xv,
                 const float* __restrict__ w0, const _Float16* __restrict__ w1,
                 const _Float16* __restrict__ w2, const _Float16* __restrict__ w3,
                 float* __restrict__ accS, float* __restrict__ accV) {
  __shared__ float    sw0[8 * 64];
  __shared__ float    sfeat[EP_WAVES][16 * 8];
  __shared__ _Float16 sA[EP_WAVES][16 * 64];
  __shared__ _Float16 sB[EP_WAVES][16 * 64];
  __shared__ int      sS[EP_WAVES][16];
  __shared__ int      sR[EP_WAVES][16];
  __shared__ float    sY[EP_WAVES][48];

  const int lane = threadIdx.x & 31;
  const int wv   = threadIdx.x >> 5;
  const int tile = blockIdx.x * EP_WAVES + wv;  // grid exactly covers E/16 tiles
  const int e0   = tile * 16;

  for (int i = threadIdx.x; i < 8 * 64; i += blockDim.x) sw0[i] = w0[i];
  for (int i = lane; i < 128; i += 32) sfeat[wv][i] = feats[(size_t)e0 * 8 + i];
  for (int i = lane; i < 48; i += 32)  sY[wv][i]    = Yg[(size_t)e0 * 3 + i];
  if (lane < 16) {
    const int sv = snd[e0 + lane];
    const int rv = rcv[e0 + lane];
    sS[wv][lane] = sv;
    sR[wv][lane] = rv;
    // Warm L2/WGP$ with the gathered rows while the MLP runs on the matrix
    // pipe (global_prefetch_b8; the gather is the memory-bound phase).
    __builtin_prefetch(Xs + (size_t)sv * CCH, 0, 1);
    if constexpr (PASS == 2) __builtin_prefetch(Xv + (size_t)sv * 3 * CCH, 0, 1);
    else                     __builtin_prefetch(accV + (size_t)rv * 3 * CCH, 1, 1);
  }
  __syncthreads();

  // layer 0: 8 -> 64 (tiny K, VALU), silu, stage f16 into LDS
  {
    const int c0 = lane, c1 = lane + 32;
#pragma unroll 4
    for (int e = 0; e < 16; ++e) {
      float a0 = 0.f, a1 = 0.f;
#pragma unroll
      for (int j = 0; j < 8; ++j) {
        const float f = sfeat[wv][e * 8 + j];
        a0 = fmaf(f, sw0[j * 64 + c0], a0);
        a1 = fmaf(f, sw0[j * 64 + c1], a1);
      }
      sA[wv][e * 64 + c0] = (_Float16)silu_f(a0);
      sA[wv][e * 64 + c1] = (_Float16)silu_f(a1);
    }
  }
  __syncthreads();
  mlp_hidden_layer(sA[wv], sB[wv], w1, lane);
  __syncthreads();
  mlp_hidden_layer(sB[wv], sA[wv], w2, lane);
  __syncthreads();

  // output layer (256 live cols) fused with message formation + scatter
  const v16h a0 = load_a_lds64(sA[wv], lane, 0);
  const v16h a1 = load_a_lds64(sA[wv], lane, 32);
  const int  rbase = (lane & 16) ? 8 : 0;
  const int  cl    = lane & 15;
  for (int t = 0; t < 16; ++t) {
    const int col = t * 16 + cl;
    v8f acc = {};
    acc = wmma16(a0, load_bp(w3, t * 2 + 0, lane), acc);
    acc = wmma16(a1, load_bp(w3, t * 2 + 1, lane), acc);
    const int chan = col & 127;
    const int path = col >> 7;  // uniform per t (no divergence around WMMA)
#pragma unroll
    for (int r = 0; r < 8; ++r) {
      const int ei = rbase + r;  // edge index within tile (C/D layout row)
      const int sn = sS[wv][ei];
      const int rc = sR[wv][ei];
      if constexpr (PASS == 1) {
        const float v = acc[r] * Xs[(size_t)sn * CCH + chan] * AVG_INV;
        if (path == 0) {
          atomicAdd(&accS[(size_t)rc * CCH + chan], v);
        } else {
          atomicAdd(&accV[((size_t)rc * 3 + 0) * CCH + chan], v * sY[wv][ei * 3 + 0]);
          atomicAdd(&accV[((size_t)rc * 3 + 1) * CCH + chan], v * sY[wv][ei * 3 + 1]);
          atomicAdd(&accV[((size_t)rc * 3 + 2) * CCH + chan], v * sY[wv][ei * 3 + 2]);
        }
      } else {
        float v;
        if (path == 0) {
          v = acc[r] * Xs[(size_t)sn * CCH + chan];
        } else {
          const float d = Xv[((size_t)sn * 3 + 0) * CCH + chan] * sY[wv][ei * 3 + 0]
                        + Xv[((size_t)sn * 3 + 1) * CCH + chan] * sY[wv][ei * 3 + 1]
                        + Xv[((size_t)sn * 3 + 2) * CCH + chan] * sY[wv][ei * 3 + 2];
          v = acc[r] * d * INV_SQRT3F;
        }
        atomicAdd(&accS[(size_t)rc * CCH + chan], v * AVG_INV);
      }
    }
  }
}

// ---------------------------------------------------------------------------
// Species-dependent (one-hot) channel mixing: per-node 128x128 matvec.
// LAYOUT 0 (skip_tp):  out[k] = sum_c in[c] * W[c][el][k]   (W: [C][EL][C])
// LAYOUT 1 (product):  out[k] = sum_c W[el][k][c] * in[c]   (W: [EL][C][C])
// ncomp = 1 (scalar) or 3 (l=1 vector; same matrix per component).
// ---------------------------------------------------------------------------
template <int LAYOUT>
__global__ void elem_mix_kernel(const float* __restrict__ in,
                                const float* __restrict__ W,
                                const int* __restrict__ species,
                                const float* __restrict__ addsrc,
                                float* __restrict__ out, int ncomp) {
  const int n  = blockIdx.x;
  const int k  = threadIdx.x;  // 0..127
  const int el = species[n];
  __shared__ float s_in[3 * CCH];
  for (int i = threadIdx.x; i < ncomp * CCH; i += blockDim.x)
    s_in[i] = in[(size_t)n * ncomp * CCH + i];
  __syncthreads();
  for (int comp = 0; comp < ncomp; ++comp) {
    float acc = 0.f;
#pragma unroll 4
    for (int c = 0; c < CCH; ++c) {
      const float w = (LAYOUT == 0) ? W[((size_t)c * NUM_EL + el) * CCH + k]
                                    : W[((size_t)el * CCH + k) * CCH + c];
      acc = fmaf(w, s_in[comp * CCH + c], acc);
    }
    const size_t o = ((size_t)n * ncomp + comp) * CCH + k;
    if (addsrc) acc += addsrc[o];
    out[o] = acc;
  }
}

// ---------------------------------------------------------------------------
// Small elementwise / per-node kernels
// ---------------------------------------------------------------------------
__global__ void edge_geom_kernel(const float* __restrict__ pos,
                                 const int* __restrict__ snd,
                                 const int* __restrict__ rcv,
                                 float* __restrict__ Yg,
                                 float* __restrict__ feats, int E) {
  const int e = blockIdx.x * blockDim.x + threadIdx.x;
  if (e >= E) return;
  const int s = snd[e], t = rcv[e];
  const float dx = pos[t * 3 + 0] - pos[s * 3 + 0];
  const float dy = pos[t * 3 + 1] - pos[s * 3 + 1];
  const float dz = pos[t * 3 + 2] - pos[s * 3 + 2];
  const float r   = sqrtf(dx * dx + dy * dy + dz * dz + 1e-12f);
  const float inv = 1.0f / r;
  Yg[e * 3 + 0] = SQRT3F * dx * inv;
  Yg[e * 3 + 1] = SQRT3F * dy * inv;
  Yg[e * 3 + 2] = SQRT3F * dz * inv;
  const float x  = r * (1.0f / RMAXF);
  const float x2 = x * x, x4 = x2 * x2, x5 = x4 * x;
  float env = 1.0f - 21.0f * x5 + 35.0f * x5 * x - 15.0f * x5 * x2;
  if (x >= 1.0f) env = 0.0f;
  const float pref = sqrtf(2.0f / RMAXF) * inv * env;
  const float w = 3.14159265358979323846f / RMAXF * r;
#pragma unroll
  for (int j = 0; j < 8; ++j)
    feats[(size_t)e * 8 + j] = pref * sinf((float)(j + 1) * w);
}

__global__ void embed_kernel(const int* __restrict__ species,
                             const float* __restrict__ Wemb,
                             float* __restrict__ h0, int Nn) {
  const int i = blockIdx.x * blockDim.x + threadIdx.x;
  if (i >= Nn * CCH) return;
  h0[i] = Wemb[species[i >> 7] * CCH + (i & 127)];
}

__global__ void readout1_kernel(const float* __restrict__ HS,
                                const float* __restrict__ ro1,
                                float* __restrict__ en, int Nn) {
  const int n = blockIdx.x * blockDim.x + threadIdx.x;
  if (n >= Nn) return;
  const float* h = HS + (size_t)n * CCH;
  float a = 0.f;
  for (int c = 0; c < CCH; ++c) a = fmaf(h[c], ro1[c], a);
  en[n] = a;
}

__global__ void readout2_kernel(const float* __restrict__ H2,
                                const float* __restrict__ roa,
                                const float* __restrict__ rob,
                                float* __restrict__ en, int Nn) {
  const int n = blockIdx.x * blockDim.x + threadIdx.x;
  if (n >= Nn) return;
  float t[16];
#pragma unroll
  for (int j = 0; j < 16; ++j) t[j] = 0.f;
  const float* h = H2 + (size_t)n * CCH;
  for (int c = 0; c < CCH; ++c) {
    const float hv = h[c];
#pragma unroll
    for (int j = 0; j < 16; ++j) t[j] = fmaf(hv, roa[c * 16 + j], t[j]);
  }
  float a = 0.f;
#pragma unroll
  for (int j = 0; j < 16; ++j) a = fmaf(silu_f(t[j]), rob[j], a);
  en[n] += a;
}

__global__ void finalize_kernel(const float* __restrict__ en,
                                const int* __restrict__ species,
                                const int* __restrict__ batch,
                                const float* __restrict__ AE,
                                const float* __restrict__ scale,
                                const float* __restrict__ shift,
                                float* __restrict__ out, int Nn) {
  const int n = blockIdx.x * blockDim.x + threadIdx.x;
  if (n >= Nn) return;
  const float v = AE[species[n]] + scale[0] * en[n] + shift[0];
  atomicAdd(&out[batch[n]], v);
}

__global__ void zero_kernel(float* __restrict__ p, size_t n) {
  const size_t i = (size_t)blockIdx.x * blockDim.x + threadIdx.x;
  if (i < n) p[i] = 0.f;
}

// ---------------------------------------------------------------------------
static inline void gemm_launch(const float* A, const _Float16* B, float* C,
                               int M, int Nn, int K, hipStream_t s) {
  const int tiles  = (M / 16) * (Nn / 16);
  const int blocks = (tiles + 7) / 8;  // 8 waves (256 threads) per block
  wmma_gemm_kernel<<<blocks, 256, 0, s>>>(A, B, C, M, Nn, K);
}

extern "C" void kernel_launch(void* const* d_in, const int* in_sizes, int n_in,
                              void* d_out, int out_size, void* d_ws,
                              size_t ws_size, hipStream_t stream) {
  (void)n_in; (void)ws_size;
  // setup_inputs() dict order, params dict flattened in insertion order
  const float* positions = (const float*)d_in[0];
  const int*   species   = (const int*)d_in[1];
  const int*   edge_idx  = (const int*)d_in[2];
  const int*   batch     = (const int*)d_in[3];
  const float* AE        = (const float*)d_in[4];
  const float* W_embed   = (const float*)d_in[5];
  const float* W_up1     = (const float*)d_in[6];
  const float* m1w0      = (const float*)d_in[7];
  const float* m1w1      = (const float*)d_in[8];
  const float* m1w2      = (const float*)d_in[9];
  const float* m1w3      = (const float*)d_in[10];
  const float* W_lin1_s  = (const float*)d_in[11];
  const float* W_lin1_v  = (const float*)d_in[12];
  const float* W_skip1_s = (const float*)d_in[13];
  const float* W_skip1_v = (const float*)d_in[14];
  const float* Wp1_s     = (const float*)d_in[15];
  const float* Wp1_v     = (const float*)d_in[16];
  const float* ro1       = (const float*)d_in[17];
  const float* W_up2_s   = (const float*)d_in[18];
  const float* W_up2_v   = (const float*)d_in[19];
  const float* m2w0      = (const float*)d_in[20];
  const float* m2w1      = (const float*)d_in[21];
  const float* m2w2      = (const float*)d_in[22];
  const float* m2w3      = (const float*)d_in[23];
  const float* W_lin2_s  = (const float*)d_in[24];
  const float* W_sc2     = (const float*)d_in[25];
  const float* Wp2_s     = (const float*)d_in[26];
  const float* ro2_a     = (const float*)d_in[27];
  const float* ro2_b     = (const float*)d_in[28];
  const float* scale     = (const float*)d_in[29];
  const float* shift     = (const float*)d_in[30];

  const int Nn = in_sizes[1];
  const int E  = in_sizes[2] / 2;
  const int G  = out_size;
  const int* snd = edge_idx;
  const int* rcv = edge_idx + E;

  // ---- workspace carve-up (f32 region then f16 region, 32B aligned) ----
  float* f = (float*)d_ws;
  size_t off = 0;
  auto take = [&](size_t n) { float* p = f + off; off += n; return p; };
  float* Yg    = take((size_t)E * 3);
  float* feats = take((size_t)E * 8);
  float* S0 = take((size_t)Nn * CCH);       // scalar ping
  float* S1 = take((size_t)Nn * CCH);       // scalar pong
  float* Sa = take((size_t)Nn * CCH);       // scalar accumulator
  float* V0 = take((size_t)Nn * 3 * CCH);   // vector ping  [N][3][C]
  float* V1 = take((size_t)Nn * 3 * CCH);   // vector pong
  float* HS = take((size_t)Nn * CCH);       // h1_s (persistent)
  float* HV = take((size_t)Nn * 3 * CCH);   // h1_v (persistent)
  float* en = take((size_t)Nn);             // per-node interaction energy
  off = (off + 7) & ~(size_t)7;             // keep f16 region 32B aligned
  _Float16* h = (_Float16*)(f + off);
  size_t hoff = 0;
  auto takeh = [&](size_t n) { _Float16* p = h + hoff; hoff += n; return p; };
  _Float16* h_up1   = takeh(CCH * CCH);
  _Float16* h_lin1s = takeh(CCH * CCH);
  _Float16* h_lin1v = takeh(CCH * CCH);
  _Float16* h_up2s  = takeh(CCH * CCH);
  _Float16* h_up2v  = takeh(CCH * CCH);
  _Float16* h_lin2s = takeh(CCH * CCH);
  _Float16* h_m1w1  = takeh(64 * 64);
  _Float16* h_m1w2  = takeh(64 * 64);
  _Float16* h_m1w3  = takeh(64 * 256);
  _Float16* h_m2w1  = takeh(64 * 64);
  _Float16* h_m2w2  = takeh(64 * 64);
  _Float16* h_m2w3  = takeh(64 * 256);  // only live 256 of 640 cols

  auto pack = [&](const float* s, _Float16* d, int K, int Ncols, int stride) {
    const int n = K * Ncols;
    pack_b_kernel<<<(n + 255) / 256, 256, 0, stream>>>(s, d, K, Ncols, stride);
  };
  auto zero = [&](float* p, size_t n) {
    zero_kernel<<<(unsigned)((n + 255) / 256), 256, 0, stream>>>(p, n);
  };

  // weights -> fragment-major f16 (every call; deterministic)
  pack(W_up1,    h_up1,   CCH, CCH, CCH);
  pack(W_lin1_s, h_lin1s, CCH, CCH, CCH);
  pack(W_lin1_v, h_lin1v, CCH, CCH, CCH);
  pack(W_up2_s,  h_up2s,  CCH, CCH, CCH);
  pack(W_up2_v,  h_up2v,  CCH, CCH, CCH);
  pack(W_lin2_s, h_lin2s, CCH, CCH, CCH);
  pack(m1w1, h_m1w1, 64, 64, 64);
  pack(m1w2, h_m1w2, 64, 64, 64);
  pack(m1w3, h_m1w3, 64, 256, 256);
  pack(m2w1, h_m2w1, 64, 64, 64);
  pack(m2w2, h_m2w2, 64, 64, 64);
  pack(m2w3, h_m2w3, 64, 256, 640);  // trim dead columns

  // ---- geometry + radial basis, node embedding ----
  edge_geom_kernel<<<(E + 255) / 256, 256, 0, stream>>>(positions, snd, rcv, Yg, feats, E);
  embed_kernel<<<(Nn * CCH + 255) / 256, 256, 0, stream>>>(species, W_embed, S0, Nn);

  // ======== interaction 1 ========
  gemm_launch(S0, h_up1, S1, Nn, CCH, CCH, stream);          // Xup1 = h0 @ W_up1
  zero(Sa, (size_t)Nn * CCH);                                 // a_s
  zero(V0, (size_t)Nn * 3 * CCH);                             // a_v
  edge_pass_kernel<1><<<E / 16 / EP_WAVES, 32 * EP_WAVES, 0, stream>>>(
      feats, Yg, snd, rcv, S1, nullptr, m1w0, h_m1w1, h_m1w2, h_m1w3, Sa, V0);
  gemm_launch(Sa, h_lin1s, S0, Nn, CCH, CCH, stream);         // a_s @ W_lin1_s
  gemm_launch(V0, h_lin1v, V1, Nn * 3, CCH, CCH, stream);     // a_v @ W_lin1_v
  elem_mix_kernel<0><<<Nn, 128, 0, stream>>>(S0, W_skip1_s, species, nullptr, Sa, 1); // m1_s
  elem_mix_kernel<0><<<Nn, 128, 0, stream>>>(V1, W_skip1_v, species, nullptr, V0, 3); // m1_v
  elem_mix_kernel<1><<<Nn, 128, 0, stream>>>(Sa, Wp1_s, species, nullptr, HS, 1);     // h1_s
  elem_mix_kernel<1><<<Nn, 128, 0, stream>>>(V0, Wp1_v, species, nullptr, HV, 3);     // h1_v
  readout1_kernel<<<(Nn + 255) / 256, 256, 0, stream>>>(HS, ro1, en, Nn);             // en1

  // ======== interaction 2 (scalar output path only; m2_v is dead) ========
  elem_mix_kernel<0><<<Nn, 128, 0, stream>>>(HS, W_sc2, species, nullptr, S0, 1);     // sc2
  gemm_launch(HS, h_up2s, S1, Nn, CCH, CCH, stream);          // Xup2_s
  gemm_launch(HV, h_up2v, V1, Nn * 3, CCH, CCH, stream);      // Xup2_v
  zero(Sa, (size_t)Nn * CCH);                                 // b_s
  edge_pass_kernel<2><<<E / 16 / EP_WAVES, 32 * EP_WAVES, 0, stream>>>(
      feats, Yg, snd, rcv, S1, V1, m2w0, h_m2w1, h_m2w2, h_m2w3, Sa, nullptr);
  gemm_launch(Sa, h_lin2s, S1, Nn, CCH, CCH, stream);         // b_s @ W_lin2_s
  elem_mix_kernel<1><<<Nn, 128, 0, stream>>>(S1, Wp2_s, species, S0, Sa, 1);          // h2_s
  readout2_kernel<<<(Nn + 255) / 256, 256, 0, stream>>>(Sa, ro2_a, ro2_b, en, Nn);    // en2

  // ======== per-graph reduction ========
  zero((float*)d_out, (size_t)G);
  finalize_kernel<<<(Nn + 255) / 256, 256, 0, stream>>>(en, species, batch, AE,
                                                        scale, shift,
                                                        (float*)d_out, Nn);
}
